// MultiHeadAttention_69801808495037
// MI455X (gfx1250) — matrix operand order repaired and checked by reading, not compiled, not run on hardware
//
#include <hip/hip_runtime.h>
#include <hip/hip_bf16.h>
#include <stdint.h>

// MHA: B=2, S=2048, D=1024, H=16, dk=64. ~70 GFLOP over ~50MB -> compute bound
// (AI > 1000 FLOP/byte; all tensors L2-resident on a 192MB L2).
// Path: bf16 WMMA (v_wmma_f32_16x16x32_bf16) with f32 accumulation, flash-style
// online softmax (never materialize the 2048x2048 score matrix).
// All operands pre-converted to bf16 once; GEMM loops are software-pipelined
// (ping-pong fragment buffers) so loads for stage n+1 fly under stage n WMMAs.

#define BATCH 2
#define SEQ   2048
#define DIM   1024
#define NHEAD 16
#define DKH   64
#define NTOK  (BATCH * SEQ)   // 4096

typedef __attribute__((ext_vector_type(16))) __bf16 v16bf;
typedef __attribute__((ext_vector_type(8)))  float  v8f;

// 16 bf16 elements per lane = 8 VGPRs; raw-bit views for loads/packing.
union Frag16 { v16bf v; uint32_t u[8]; uint4 q[2]; };

__device__ __forceinline__ uint32_t pack2_bf16(float a, float b) {
  uint32_t ua = __float_as_uint(a);
  uint32_t ub = __float_as_uint(b);
  ua = (ua + 0x7FFFu + ((ua >> 16) & 1u)) >> 16;   // round-to-nearest-even
  ub = (ub + 0x7FFFu + ((ub >> 16) & 1u)) >> 16;
  return ua | (ub << 16);
}

__device__ __forceinline__ uint16_t f32_to_bf16(float a) {
  uint32_t ua = __float_as_uint(a);
  return (uint16_t)((ua + 0x7FFFu + ((ua >> 16) & 1u)) >> 16);
}

__device__ __forceinline__ v8f wmma_bf16(const Frag16& a, const Frag16& b, v8f c) {
  // (neg_a, A, neg_b, B, c_mod, C, reuse_a, reuse_b)
  return __builtin_amdgcn_wmma_f32_16x16x32_bf16(false, a.v, false, b.v,
                                                 (short)0, c, false, false);
}

// ---------------------------------------------------------------------------
// Kernel 0a: one-pass fp32 -> bf16 conversion of x.
// ---------------------------------------------------------------------------
__global__ __launch_bounds__(256)
void k_cvt_x(const float* __restrict__ x, uint16_t* __restrict__ xbf) {
  const size_t i = ((size_t)blockIdx.x * 256 + threadIdx.x) * 8;
  const float4 f0 = *(const float4*)(x + i);
  const float4 f1 = *(const float4*)(x + i + 4);
  uint4 o;
  o.x = pack2_bf16(f0.x, f0.y);
  o.y = pack2_bf16(f0.z, f0.w);
  o.z = pack2_bf16(f1.x, f1.y);
  o.w = pack2_bf16(f1.z, f1.w);
  *(uint4*)(xbf + i) = o;
}

// ---------------------------------------------------------------------------
// Kernel 0b: transpose 4 weight matrices to bf16, column-major:
// wt[z][n][k] = bf16(W_z[k][n]) -> every B-fragment is a contiguous 32B load.
// ---------------------------------------------------------------------------
__global__ __launch_bounds__(256)
void k_transpose(const float* __restrict__ Wq, const float* __restrict__ Wk,
                 const float* __restrict__ Wv, const float* __restrict__ Wo,
                 uint16_t* __restrict__ wt) {
  __shared__ float tile[16][17];
  const int z = blockIdx.z;
  const float* W = (z == 0) ? Wq : (z == 1) ? Wk : (z == 2) ? Wv : Wo;
  uint16_t* dst = wt + (size_t)z * DIM * DIM;
  const int tx = threadIdx.x & 15, ty = threadIdx.x >> 4;
  const int k = blockIdx.y * 16 + ty;
  const int n = blockIdx.x * 16 + tx;
  tile[ty][tx] = W[(size_t)k * DIM + n];
  __syncthreads();
  const int nn = blockIdx.x * 16 + ty;
  const int kk = blockIdx.y * 16 + tx;
  dst[(size_t)nn * DIM + kk] = f32_to_bf16(tile[tx][ty]);
}

// ---------------------------------------------------------------------------
// Shared GEMM-strip machinery: one wave computes a 16-row x 64-col strip of
// A[rows x DIM](bf16) * Bt[cols x DIM](bf16, pre-transposed), software-
// pipelined with ping-pong fragment buffers (step 64 in K = 2 stages).
// ---------------------------------------------------------------------------
__device__ __forceinline__ void load_afrag(const uint16_t* __restrict__ arow,
                                           int k0, int hi, Frag16& a) {
  a.q[0] = *(const uint4*)(arow + k0 + hi * 16);
  a.q[1] = *(const uint4*)(arow + k0 + hi * 16 + 8);
}

__device__ __forceinline__ void load_bfrags(const uint16_t* __restrict__ wt,
                                            int col0, int lo, int hi, int k0,
                                            Frag16* b4) {
#pragma unroll
  for (int j = 0; j < 4; ++j) {
    const uint4* wp =
        (const uint4*)(wt + (size_t)(col0 + j * 16 + lo) * DIM + k0 + hi * 16);
    b4[j].q[0] = wp[0];
    b4[j].q[1] = wp[1];
  }
}

__device__ __forceinline__ void gemm_strip(const uint16_t* __restrict__ arow,
                                           const uint16_t* __restrict__ wt,
                                           int col0, int lo, int hi, v8f acc[4]) {
  Frag16 aA, aB, bA[4], bB[4];
  load_afrag(arow, 0, hi, aA);
  load_bfrags(wt, col0, lo, hi, 0, bA);
  for (int k0 = 0; k0 < DIM; k0 += 64) {
    // stage 0: prefetch k0+32 into B-buffers, compute with A-buffers
    load_afrag(arow, k0 + 32, hi, aB);
    load_bfrags(wt, col0, lo, hi, k0 + 32, bB);
    __builtin_prefetch((const void*)(arow + k0 + 128), 0, 3);
#pragma unroll
    for (int j = 0; j < 4; ++j) acc[j] = wmma_bf16(aA, bA[j], acc[j]);
    // stage 1: prefetch k0+64 (clamped; last reload is harmless), compute
    const int kn = (k0 + 64 < DIM) ? (k0 + 64) : 0;
    load_afrag(arow, kn, hi, aA);
    load_bfrags(wt, col0, lo, hi, kn, bA);
#pragma unroll
    for (int j = 0; j < 4; ++j) acc[j] = wmma_bf16(aB, bB[j], acc[j]);
  }
}

// ---------------------------------------------------------------------------
// Kernel 1: fused QKV projection.  grid = (NTOK/128, DIM/64, 3), 8 waves/block.
// Stores: Q (x 1/sqrt(dk)) and K as bf16 [b,h,s,dk]; V transposed [b,h,dk,s].
// ---------------------------------------------------------------------------
__global__ __launch_bounds__(256)
void k_qkv(const uint16_t* __restrict__ xbf, const uint16_t* __restrict__ wt,
           const float* __restrict__ bq, const float* __restrict__ bk,
           const float* __restrict__ bv,
           uint16_t* __restrict__ qbf, uint16_t* __restrict__ kbf,
           uint16_t* __restrict__ vtbf) {
  const int lane = threadIdx.x & 31;
  const int wv   = threadIdx.x >> 5;
  const int lo = lane & 15, hi = lane >> 4;
  const int row0 = blockIdx.x * 128 + wv * 16;
  const int col0 = blockIdx.y * 64;
  const int z = blockIdx.z;
  const uint16_t* w = wt + (size_t)z * DIM * DIM;

  v8f acc[4];
#pragma unroll
  for (int j = 0; j < 4; ++j) acc[j] = {};

  const uint16_t* xrow = xbf + (size_t)(row0 + lo) * DIM;
  gemm_strip(xrow, w, col0, lo, hi, acc);

  const float* bias = (z == 0) ? bq : (z == 1) ? bk : bv;
#pragma unroll
  for (int j = 0; j < 4; ++j) {
    const int col = col0 + j * 16 + lo;
    const int h = col >> 6;   // head (== blockIdx.y)
    const int d = col & 63;
    const float bb = bias[col];
#pragma unroll
    for (int r = 0; r < 8; ++r) {
      const int tok = row0 + hi * 8 + r;     // C layout: M = r + 8*hi
      const int b = tok >> 11;               // tok / SEQ
      const int s = tok & (SEQ - 1);
      const float val = acc[j][r] + bb;
      if (z == 0)
        qbf[((size_t)(b * NHEAD + h) * SEQ + s) * DKH + d] =
            f32_to_bf16(val * 0.125f);       // fold 1/sqrt(64)
      else if (z == 1)
        kbf[((size_t)(b * NHEAD + h) * SEQ + s) * DKH + d] = f32_to_bf16(val);
      else
        vtbf[((size_t)(b * NHEAD + h) * DKH + d) * SEQ + s] = f32_to_bf16(val);
    }
  }
}

// ---------------------------------------------------------------------------
// Kernel 2: flash attention.  grid = B*H*(S/128), 8 waves/block; each wave owns
// a 16-query tile of one (b,h) and streams all 2048 keys in 32-key blocks.
// Scores computed TRANSPOSED (Sᵀ = K_tile·Qᵀ) so the C layout (lane=query,
// VGPR=key) converts to the P A-fragment with no cross-lane traffic.
// V fragments are loaded before the softmax so exp/pack VALU hides their
// latency ahead of the P*V WMMAs.
// ---------------------------------------------------------------------------
__global__ __launch_bounds__(256)
void k_attn(const uint16_t* __restrict__ qbf, const uint16_t* __restrict__ kbf,
            const uint16_t* __restrict__ vtbf, uint16_t* __restrict__ attnbf) {
  const int lane = threadIdx.x & 31, wv = threadIdx.x >> 5;
  const int lo = lane & 15, hi = lane >> 4;
  const int bh = blockIdx.x >> 4;
  const int q0 = (blockIdx.x & 15) * 128 + wv * 16;
  const int b = bh >> 4, h = bh & 15;

  // Q as B-matrix (Qᵀ): lane n = query lo, K(dk) = hi*16..+15 contiguous.
  Frag16 qb0, qb1;
  {
    const uint16_t* qp = qbf + ((size_t)bh * SEQ + q0 + lo) * DKH;
    const uint4* p0 = (const uint4*)(qp + hi * 16);
    const uint4* p1 = (const uint4*)(qp + 32 + hi * 16);
    qb0.q[0] = p0[0]; qb0.q[1] = p0[1];
    qb1.q[0] = p1[0]; qb1.q[1] = p1[1];
  }

  v8f o[4];
#pragma unroll
  for (int t = 0; t < 4; ++t) o[t] = {};
  const v8f vzero = {};
  float Mrun = -1e30f, Lrun = 0.0f;

  for (int kt = 0; kt < SEQ; kt += 32) {
    // ---- scores (transposed): two 16-key tiles ----
    v8f c[2];
#pragma unroll
    for (int half = 0; half < 2; ++half) {
      const int key = kt + half * 16 + lo;
      const uint16_t* kp = kbf + ((size_t)bh * SEQ + key) * DKH;
      Frag16 a0, a1;                          // K rows as A (16 keys x 32 dk)
      a0.q[0] = *(const uint4*)(kp + hi * 16);
      a0.q[1] = *(const uint4*)(kp + hi * 16 + 8);
      a1.q[0] = *(const uint4*)(kp + 32 + hi * 16);
      a1.q[1] = *(const uint4*)(kp + 40 + hi * 16);
      v8f s = wmma_bf16(a0, qb0, vzero);      // dk 0..31
      s = wmma_bf16(a1, qb1, s);              // dk 32..63
      c[half] = s;                            // [keys x queries], lane col = lo
    }

    // ---- V fragments early: their latency is hidden by the softmax VALU ----
    Frag16 vf[4];
#pragma unroll
    for (int t = 0; t < 4; ++t) {
      // V as B (32 keys x 16 dk): lane d = t*16+lo, keys kt+hi*8..+7 and kt+16+hi*8..+7.
      const uint4* vp = (const uint4*)(vtbf + ((size_t)bh * DKH + t * 16 + lo) * SEQ +
                                       kt + hi * 8);
      vf[t].q[0] = vp[0];
      vf[t].q[1] = vp[2];
    }
    __builtin_prefetch((const void*)(kbf + ((size_t)bh * SEQ + kt + 32 + lo) * DKH),
                       0, 3);                 // next key tile

    // ---- online softmax per query column (lane & lane^16 = 16 key rows) ----
    float tmax = c[0][0];
#pragma unroll
    for (int r = 0; r < 8; ++r) {
      tmax = fmaxf(tmax, c[0][r]);
      tmax = fmaxf(tmax, c[1][r]);
    }
    tmax = fmaxf(tmax, __shfl_xor(tmax, 16, 32));
    const float newM = fmaxf(Mrun, tmax);
    const float alpha = __expf(Mrun - newM);

    float tsum = 0.0f;
    Frag16 pa;                                // P as A fragment (16q x 32keys)
#pragma unroll
    for (int r = 0; r < 4; ++r) {
      const float p0 = __expf(c[0][2 * r]     - newM);
      const float p1 = __expf(c[0][2 * r + 1] - newM);
      const float p2 = __expf(c[1][2 * r]     - newM);
      const float p3 = __expf(c[1][2 * r + 1] - newM);
      tsum += p0 + p1 + p2 + p3;
      pa.u[r]     = pack2_bf16(p0, p1);       // keys 0..7  / 8..15 (by half)
      pa.u[4 + r] = pack2_bf16(p2, p3);       // keys 16..23 / 24..31
    }
    tsum += __shfl_xor(tsum, 16, 32);
    Lrun = Lrun * alpha + tsum;
    Mrun = newM;

    // Broadcast per-row rescale factors (row q lives in lane q / q+16).
    float arow[8];
#pragma unroll
    for (int r = 0; r < 8; ++r) arow[r] = __shfl(alpha, hi * 8 + r, 32);

#pragma unroll
    for (int t = 0; t < 4; ++t) {
#pragma unroll
      for (int r = 0; r < 8; ++r) o[t][r] *= arow[r];
      o[t] = wmma_bf16(pa, vf[t], o[t]);
    }
  }

  const float rinv = 1.0f / Lrun;
  float lrow[8];
#pragma unroll
  for (int r = 0; r < 8; ++r) lrow[r] = __shfl(rinv, hi * 8 + r, 32);

#pragma unroll
  for (int t = 0; t < 4; ++t) {
#pragma unroll
    for (int r = 0; r < 8; ++r) {
      const int s = q0 + hi * 8 + r;
      attnbf[(size_t)(b * SEQ + s) * DIM + h * 64 + t * 16 + lo] =
          f32_to_bf16(o[t][r] * lrow[r]);     // token-major for out projection
    }
  }
}

// ---------------------------------------------------------------------------
// Kernel 3: output projection (bf16 GEMM + bias -> f32).
// ---------------------------------------------------------------------------
__global__ __launch_bounds__(256)
void k_outproj(const uint16_t* __restrict__ attnbf, const uint16_t* __restrict__ wto,
               const float* __restrict__ bo, float* __restrict__ out) {
  const int lane = threadIdx.x & 31, wv = threadIdx.x >> 5;
  const int lo = lane & 15, hi = lane >> 4;
  const int row0 = blockIdx.x * 128 + wv * 16;
  const int col0 = blockIdx.y * 64;

  v8f acc[4];
#pragma unroll
  for (int j = 0; j < 4; ++j) acc[j] = {};

  const uint16_t* arow = attnbf + (size_t)(row0 + lo) * DIM;
  gemm_strip(arow, wto, col0, lo, hi, acc);

#pragma unroll
  for (int j = 0; j < 4; ++j) {
    const int col = col0 + j * 16 + lo;
    const float bb = bo[col];
#pragma unroll
    for (int r = 0; r < 8; ++r) {
      const int tok = row0 + hi * 8 + r;
      out[(size_t)tok * DIM + col] = acc[j][r] + bb;
    }
  }
}

// ---------------------------------------------------------------------------
extern "C" void kernel_launch(void* const* d_in, const int* in_sizes, int n_in,
                              void* d_out, int out_size, void* d_ws, size_t ws_size,
                              hipStream_t stream) {
  const float* x  = (const float*)d_in[0];
  const float* Wq = (const float*)d_in[1];
  const float* bq = (const float*)d_in[2];
  const float* Wk = (const float*)d_in[3];
  const float* bk = (const float*)d_in[4];
  const float* Wv = (const float*)d_in[5];
  const float* bv = (const float*)d_in[6];
  const float* Wo = (const float*)d_in[7];
  const float* bo = (const float*)d_in[8];
  float* out = (float*)d_out;

  // Workspace (bf16 raw bits): x, Q, K, Vᵀ, attn_out (8 MiB each) + 4
  // transposed weights (2 MiB each) = 48 MiB total.
  uint16_t* ws = (uint16_t*)d_ws;
  const size_t MAT = (size_t)NTOK * DIM;   // 4,194,304 elements
  uint16_t* xbf    = ws;
  uint16_t* qbf    = xbf + MAT;
  uint16_t* kbf    = qbf + MAT;
  uint16_t* vtbf   = kbf + MAT;
  uint16_t* attnbf = vtbf + MAT;
  uint16_t* wt     = attnbf + MAT;         // [4][DIM][DIM]

  k_cvt_x<<<dim3(NTOK * DIM / (8 * 256)), 256, 0, stream>>>(x, xbf);
  k_transpose<<<dim3(DIM / 16, DIM / 16, 4), 256, 0, stream>>>(Wq, Wk, Wv, Wo, wt);
  k_qkv<<<dim3(NTOK / 128, DIM / 64, 3), 256, 0, stream>>>(xbf, wt, bq, bk, bv,
                                                           qbf, kbf, vtbf);
  k_attn<<<dim3(BATCH * NHEAD * (SEQ / 128)), 256, 0, stream>>>(qbf, kbf, vtbf,
                                                                attnbf);
  k_outproj<<<dim3(NTOK / 128, DIM / 64), 256, 0, stream>>>(
      attnbf, wt + (size_t)3 * DIM * DIM, bo, out);
}
